// MiniCausalAttention_1125281431977
// MI455X (gfx1250) — matrix operand order repaired and checked by reading, compiled, not run on hardware
//
#include <hip/hip_runtime.h>

// ---------------------------------------------------------------------------
// MiniCausalAttention for MI455X (gfx1250, wave32, WMMA + async global->LDS)
//   B=4, L=2048, D=1024, fp32 in/out, bf16 WMMA internals (f32 accumulate)
// Pipeline:
//   1) x -> bf16                        (cvt kernel)
//   2) [bq|bk|bv] -> bqkv               (concat kernel)
//   3) wq/wk/wv -> combined W^T bf16    (transpose kernel, N=3072 stacked)
//   4) qkv = x@[Wq|Wk|Wv] + bqkv        (ONE WMMA GEMM, N=3072, bf16 out)
//   5) V slice -> V^T bf16              (transpose kernel, ldin=3072)
//   6) S = Q K^T / 32, causal masked    (WMMA GEMM, f32 out, upper tiles skipped)
//   7) P = softmax_row(S)               (wave-per-row, row in registers, bf16 out)
//   8) out = P V                        (WMMA GEMM, f32 out, k-limit = query block)
// GEMM mainloop: 128x256x32 block, 64x64 wave tile (16 WMMAs : 16 ds_load_b128
// per K-step = 1.0 LDS read/WMMA), double-buffered LDS fed by
// global_load_async_to_lds_b128 (ASYNCcnt), one barrier per K-step.
// ---------------------------------------------------------------------------

typedef unsigned short u16;
typedef unsigned int   u32;

typedef __attribute__((ext_vector_type(16))) __bf16 bf16x16;
typedef __attribute__((ext_vector_type(8)))  float  f32x8;

union Frag16 {           // 16 bf16 elements == 8 VGPRs
    bf16x16 v;
    uint4   q[2];
};

struct alignas(8) U16x4 { u16 x, y, z, w; };

__device__ __forceinline__ u16 f32_to_bf16_u(float f) {
    u32 x = __float_as_uint(f);
    u32 r = x + 0x7FFFu + ((x >> 16) & 1u);   // round-to-nearest-even
    return (u16)(r >> 16);
}
__device__ __forceinline__ float bf16_to_f32(u16 h) {
    return __uint_as_float(((u32)h) << 16);
}
__device__ __forceinline__ float to_f32(float x) { return x; }
__device__ __forceinline__ float to_f32(u16 x)   { return bf16_to_f32(x); }

// async copy of 16B from global to LDS (GV mode, tracked by ASYNCcnt)
__device__ __forceinline__ void async_b128(u32 lds_byte, const u16* gaddr) {
    asm volatile("global_load_async_to_lds_b128 %0, %1, off"
                 :: "v"(lds_byte), "v"(gaddr) : "memory");
}
__device__ __forceinline__ void wait_async0() {
#if __has_builtin(__builtin_amdgcn_s_wait_asynccnt)
    __builtin_amdgcn_s_wait_asynccnt(0);
#else
    asm volatile("s_wait_asynccnt 0" ::: "memory");
#endif
}

// ---------------------------------------------------------------------------
// 1) elementwise f32 -> bf16 (vec4)
// ---------------------------------------------------------------------------
__global__ __launch_bounds__(256) void cvt_f32_to_bf16(
    const float* __restrict__ in, u16* __restrict__ out, long n)
{
    long i = ((long)blockIdx.x * blockDim.x + threadIdx.x) * 4;
    if (i >= n) return;
    float4 f = *(const float4*)(in + i);
    U16x4 o;
    o.x = f32_to_bf16_u(f.x); o.y = f32_to_bf16_u(f.y);
    o.z = f32_to_bf16_u(f.z); o.w = f32_to_bf16_u(f.w);
    *(U16x4*)(out + i) = o;
}

// ---------------------------------------------------------------------------
// 2) concat three 1024-float bias vectors into one 3072-float buffer
// ---------------------------------------------------------------------------
__global__ __launch_bounds__(256) void concat_bias3(
    const float* __restrict__ a, const float* __restrict__ b,
    const float* __restrict__ c, float* __restrict__ o)
{
    int i = blockIdx.x * 256 + threadIdx.x;        // grid 12 -> 3072
    float v = (i < 1024) ? a[i] : (i < 2048 ? b[i - 1024] : c[i - 2048]);
    o[i] = v;
}

// ---------------------------------------------------------------------------
// 3/5) batched 32x32 tiled transpose, any-input -> bf16.
//      out[c*R + r] = in[r*ldin + c]   (out packed [C][R])
// grid: (C/32, R/32, batch)   block: 256 (32x8)
// ---------------------------------------------------------------------------
template <typename Tin>
__global__ __launch_bounds__(256) void transpose_to_bf16(
    const Tin* __restrict__ in, u16* __restrict__ out,
    int R, int C, int ldin, long sIn, long sOut)
{
    __shared__ float tile[32][33];
    in  += (long)blockIdx.z * sIn;
    out += (long)blockIdx.z * sOut;
    int c0 = blockIdx.x * 32, r0 = blockIdx.y * 32;
    int tx = threadIdx.x & 31, ty = threadIdx.x >> 5;   // ty in 0..7
#pragma unroll
    for (int i = 0; i < 32; i += 8)
        tile[ty + i][tx] = to_f32(in[(long)(r0 + ty + i) * ldin + c0 + tx]);
    __syncthreads();
#pragma unroll
    for (int i = 0; i < 32; i += 8)
        out[(long)(c0 + ty + i) * R + r0 + tx] = f32_to_bf16_u(tile[tx][ty + i]);
}

// ---------------------------------------------------------------------------
// WMMA GEMM:  C[M,N] = scale * (A[M,K] @ Bt[N,K]^T) + bias[N]
//   A, Bt are bf16 row-major (Bt stores B transposed: fixed n => contiguous k,
//   matching the per-lane WMMA B-fragment access pattern).
//   BLOCK 128x256x32, 256 threads = 8 waves (2x4), wave tile 64x64 (4x4 WMMAs).
//   LDS double-buffered; each K-step: async DMA of next 128x32 A and 256x32 B
//   slabs into buffer^1 (6x b128/thread), 16 WMMAs from current buffer,
//   s_wait_asynccnt 0 + one barrier.
//   CMASK : causal -inf mask on outputs (score kernel); fully-masked tiles skipped.
//   CKLIM : k-loop stops at mblock end (P@V under causality).
// grid: (M/128, N/256, batch)
// ---------------------------------------------------------------------------
#define LDSB_A 10240   // bytes per A buffer: 128 rows * 40 u16 * 2B
#define LDSB_B 20480   // bytes per B buffer: 256 rows * 40 u16 * 2B

template <typename OUT, bool CMASK, bool CKLIM>
__global__ __launch_bounds__(256, 1) void gemm_wmma_bf16(
    const u16* __restrict__ A, const u16* __restrict__ Bt,
    const float* __restrict__ bias, OUT* __restrict__ C,
    int M, int N, int K, int lda, int ldb, int ldc,
    long sA, long sB, long sC, float scale)
{
    const int mbase = blockIdx.x * 128;
    const int nbase = blockIdx.y * 256;
    if (CMASK && nbase > mbase + 127) return;          // tile entirely above diagonal

    A  += (long)blockIdx.z * sA;
    Bt += (long)blockIdx.z * sB;
    C  += (long)blockIdx.z * sC;

    const int kend = CKLIM ? (mbase + 128 < K ? mbase + 128 : K) : K;

    // double-buffered LDS tiles, row stride 40 u16 (80B) => conflict-free b128 reads
    __shared__ u16 As[2][128 * 40];
    __shared__ u16 Bs[2][256 * 40];

    const int tid  = threadIdx.x;
    const int lane = tid & 31;
    const int w    = tid >> 5;
    const int wm   = w >> 2;          // 0..1 : 64-row slab
    const int wn   = w & 3;           // 0..3 : 64-col slab
    const int half = lane >> 4;       // hi/lo 16 lanes
    const int mrow = lane & 15;

    // per-thread DMA chunks (16B each): A rows (r, r+64); B rows (r, r+64, r+128, r+192)
    const int row0 = tid >> 2;        // 0..63
    const int seg  = tid & 3;         // 0..3 (16B segment within a 64B row)
    const u16* gA0 = A  + (long)(mbase + row0) * lda + seg * 8;
    const u16* gA1 = gA0 + (long)64 * lda;
    const u16* gB0 = Bt + (long)(nbase + row0) * ldb + seg * 8;
    const u16* gB1 = gB0 + (long)64  * ldb;
    const u16* gB2 = gB0 + (long)128 * ldb;
    const u16* gB3 = gB0 + (long)192 * ldb;

    const u32 asb  = (u32)(size_t)&As[0][0];
    const u32 bsb  = (u32)(size_t)&Bs[0][0];
    const u32 offL0 = (u32)(row0 * 80 + seg * 16);     // LDS byte offset, row r
    const u32 rowblk = 64u * 80u;                      // 64 rows of LDS

    f32x8 acc[4][4] = {};             // 4 M-tiles x 4 N-tiles of 16x16 f32

    const int nsteps = kend >> 5;     // kend is a multiple of 32

    // prologue: DMA first K-slab into buffer 0
    async_b128(asb + offL0,              gA0);
    async_b128(asb + offL0 + rowblk,     gA1);
    async_b128(bsb + offL0,              gB0);
    async_b128(bsb + offL0 + rowblk,     gB1);
    async_b128(bsb + offL0 + 2 * rowblk, gB2);
    async_b128(bsb + offL0 + 3 * rowblk, gB3);
    gA0 += 32; gA1 += 32; gB0 += 32; gB1 += 32; gB2 += 32; gB3 += 32;
    wait_async0();
    __syncthreads();

    for (int it = 0; it < nsteps; ++it) {
        const int cur = it & 1;

        if (it + 1 < nsteps) {        // uniform: overlap DMA of next slab with WMMA
            const u32 na = asb + (u32)((cur ^ 1) * LDSB_A) + offL0;
            const u32 nb = bsb + (u32)((cur ^ 1) * LDSB_B) + offL0;
            async_b128(na,              gA0);
            async_b128(na + rowblk,     gA1);
            async_b128(nb,              gB0);
            async_b128(nb + rowblk,     gB1);
            async_b128(nb + 2 * rowblk, gB2);
            async_b128(nb + 3 * rowblk, gB3);
            gA0 += 32; gA1 += 32; gB0 += 32; gB1 += 32; gB2 += 32; gB3 += 32;
        }

        const u16* as = &As[cur][0];
        const u16* bs = &Bs[cur][0];

        // B fragments: lane n = mrow, 16 contiguous k per half-wave
        Frag16 bfr[4];
#pragma unroll
        for (int nt = 0; nt < 4; ++nt) {
            const u16* pb = &bs[(wn * 64 + nt * 16 + mrow) * 40 + (half << 4)];
            bfr[nt].q[0] = *(const uint4*)pb;
            bfr[nt].q[1] = *(const uint4*)(pb + 8);
        }
#pragma unroll
        for (int mi = 0; mi < 4; ++mi) {
            // A fragment: row m = mrow; elems 0..15 = K half*16 .. half*16+15
            Frag16 af;
            const u16* pa = &as[(wm * 64 + mi * 16 + mrow) * 40];
            af.q[0] = *(const uint4*)(pa + (half << 4));
            af.q[1] = *(const uint4*)(pa + (half << 4) + 8);
#pragma unroll
            for (int nt = 0; nt < 4; ++nt) {
                acc[mi][nt] = __builtin_amdgcn_wmma_f32_16x16x32_bf16(
                    false, af.v, false, bfr[nt].v, (short)0, acc[mi][nt], false, false);
            }
        }

        wait_async0();                // next buffer fully written
        __syncthreads();              // all waves done reading current buffer
    }

    // epilogue: C/D layout => VGPR r: (row r, lanes 0-15) / (row r+8, lanes 16-31)
#pragma unroll
    for (int mi = 0; mi < 4; ++mi) {
#pragma unroll
        for (int nt = 0; nt < 4; ++nt) {
            const int n  = nbase + wn * 64 + nt * 16 + mrow;
            const float bv = bias ? bias[n] : 0.0f;
            f32x8 v = acc[mi][nt];
#pragma unroll
            for (int r = 0; r < 8; ++r) {
                const int m = mbase + wm * 64 + mi * 16 + r + half * 8;
                float f = v[r] * scale + bv;
                if (CMASK && n > m) f = -__builtin_inff();
                if constexpr (__is_same(OUT, float))
                    C[(long)m * ldc + n] = f;
                else
                    C[(long)m * ldc + n] = f32_to_bf16_u(f);
            }
        }
    }
}

// ---------------------------------------------------------------------------
// 7) causal row softmax, row resident in registers.
//    One wave per row; S f32 [B*L, L] -> P bf16 [B*L, L].
//    Only the first ni = (qi>>7)+1 chunks of 128 cols are read/written; the
//    P@V GEMM's causal k-limit never reads beyond that 128-block boundary.
// grid: (B*L/8), block 256 (8 waves)
// ---------------------------------------------------------------------------
__global__ __launch_bounds__(256) void softmax_causal(
    const float* __restrict__ S, u16* __restrict__ P, int L)
{
    const int w = threadIdx.x >> 5, lane = threadIdx.x & 31;
    const long row = (long)blockIdx.x * 8 + w;
    const int qi = (int)(row % L);                 // valid columns: 0..qi
    const float* s = S + row * (long)L;
    u16* p = P + row * (long)L;
    const int ni = (qi >> 7) + 1;                  // 128-col chunks (<= 16)

    float4 regs[16];
    float mx = -3.4e38f;
#pragma unroll
    for (int i = 0; i < 16; ++i) {
        if (i < ni) {
            const int c0 = lane * 4 + i * 128;
            float4 f = *(const float4*)(s + c0);
            // mask by replacement (NaN-safe; exp of -3.4e38 underflows to 0)
            if (c0 + 0 > qi) f.x = -3.4e38f;
            if (c0 + 1 > qi) f.y = -3.4e38f;
            if (c0 + 2 > qi) f.z = -3.4e38f;
            if (c0 + 3 > qi) f.w = -3.4e38f;
            regs[i] = f;
            mx = fmaxf(mx, fmaxf(fmaxf(f.x, f.y), fmaxf(f.z, f.w)));
        }
    }
#pragma unroll
    for (int o = 16; o >= 1; o >>= 1) mx = fmaxf(mx, __shfl_xor(mx, o, 32));

    float sum = 0.0f;
#pragma unroll
    for (int i = 0; i < 16; ++i) {
        if (i < ni) {
            float4 f = regs[i];
            f.x = __expf(f.x - mx); f.y = __expf(f.y - mx);
            f.z = __expf(f.z - mx); f.w = __expf(f.w - mx);
            regs[i] = f;
            sum += (f.x + f.y) + (f.z + f.w);
        }
    }
#pragma unroll
    for (int o = 16; o >= 1; o >>= 1) sum += __shfl_xor(sum, o, 32);
    const float inv = 1.0f / sum;

#pragma unroll
    for (int i = 0; i < 16; ++i) {
        if (i < ni) {
            const int c0 = lane * 4 + i * 128;
            float4 f = regs[i];
            U16x4 o;
            o.x = f32_to_bf16_u(f.x * inv);
            o.y = f32_to_bf16_u(f.y * inv);
            o.z = f32_to_bf16_u(f.z * inv);
            o.w = f32_to_bf16_u(f.w * inv);
            *(U16x4*)(p + c0) = o;
        }
    }
}

// ---------------------------------------------------------------------------
// host launcher
// ---------------------------------------------------------------------------
extern "C" void kernel_launch(void* const* d_in, const int* in_sizes, int n_in,
                              void* d_out, int out_size, void* d_ws, size_t ws_size,
                              hipStream_t stream)
{
    (void)in_sizes; (void)n_in; (void)out_size; (void)ws_size;
    constexpr int B = 4, L = 2048, D = 1024;
    constexpr int M = B * L;                    // 8192 query rows total
    constexpr int N3 = 3 * D;                   // 3072: stacked Q|K|V columns

    const float* x  = (const float*)d_in[0];
    const float* wq = (const float*)d_in[1];
    const float* bq = (const float*)d_in[2];
    const float* wk = (const float*)d_in[3];
    const float* bk = (const float*)d_in[4];
    const float* wv = (const float*)d_in[5];
    const float* bv = (const float*)d_in[6];
    float* out = (float*)d_out;

    // workspace carve-up (256B aligned)
    char* ws = (char*)d_ws;
    auto carve = [&](size_t bytes) -> char* {
        char* p = ws;
        ws += (bytes + 255) & ~(size_t)255;
        return p;
    };
    u16*   xb   = (u16*)carve((size_t)M * D * 2);          // x bf16
    u16*   wt   = (u16*)carve((size_t)N3 * D * 2);         // [Wq|Wk|Wv]^T bf16 [3072][1024]
    float* bqkv = (float*)carve((size_t)N3 * 4);           // stacked bias
    u16*   qkv  = (u16*)carve((size_t)M * N3 * 2);         // interleaved Q|K|V bf16 [M][3072]
    u16*   vtb  = (u16*)carve((size_t)M * D * 2);          // V^T bf16 (per batch [D][L])
    float* Sb   = (float*)carve((size_t)B * L * L * 4);    // scores f32
    u16*   Pb   = (u16*)carve((size_t)B * L * L * 2);      // probs bf16

    const u16* qb = qkv;            // column slices of the interleaved buffer
    const u16* kb = qkv + D;
    const u16* vb = qkv + 2 * D;

    // 1) x -> bf16
    {
        long n = (long)M * D;
        cvt_f32_to_bf16<<<dim3((unsigned)(n / 4 / 256)), dim3(256), 0, stream>>>(x, xb, n);
    }
    // 2) bias concat
    concat_bias3<<<dim3(N3 / 256), dim3(256), 0, stream>>>(bq, bk, bv, bqkv);
    // 3) weights -> stacked W^T bf16 [3072][1024]
    {
        dim3 g(D / 32, D / 32, 1), blk(256);
        transpose_to_bf16<float><<<g, blk, 0, stream>>>(wq, wt,             D, D, D, 0, 0);
        transpose_to_bf16<float><<<g, blk, 0, stream>>>(wk, wt + D * D,     D, D, D, 0, 0);
        transpose_to_bf16<float><<<g, blk, 0, stream>>>(wv, wt + 2 * D * D, D, D, D, 0, 0);
    }
    // 4) fused QKV projection: [8192,1024] @ [1024,3072] + bias, bf16 out (one launch)
    {
        dim3 g(M / 128, N3 / 256, 1), blk(256);
        gemm_wmma_bf16<u16, false, false><<<g, blk, 0, stream>>>(
            xb, wt, bqkv, qkv, M, N3, D, D, D, N3, 0, 0, 0, 1.0f);
    }
    // 5) V slice -> V^T (per batch [L,3072-strided] -> [D,L])
    {
        dim3 g(D / 32, L / 32, B), blk(256);
        transpose_to_bf16<u16><<<g, blk, 0, stream>>>(
            vb, vtb, L, D, N3, (long)L * N3, (long)D * L);
    }
    // 6) S = Q K^T / sqrt(D), causal mask (K slice [L][D] @ stride 3072 is Bt layout)
    {
        dim3 g(L / 128, L / 256, B), blk(256);
        gemm_wmma_bf16<float, true, false><<<g, blk, 0, stream>>>(
            qb, kb, nullptr, Sb, L, L, D, N3, N3, L,
            (long)L * N3, (long)L * N3, (long)L * L, 1.0f / 32.0f);
    }
    // 7) P = softmax(S) row-wise, bf16, only up to causal 128-block boundary
    {
        softmax_causal<<<dim3(M / 8), dim3(256), 0, stream>>>(Sb, Pb, L);
    }
    // 8) out = P @ V : [L,L] @ [L,D] per batch, k-limit from causality, f32 out
    {
        dim3 g(L / 128, D / 256, B), blk(256);
        gemm_wmma_bf16<float, false, true><<<g, blk, 0, stream>>>(
            Pb, vtb, nullptr, out, L, D, L, L, L, D,
            (long)L * L, (long)D * L, (long)L * D, 1.0f);
    }
}
